// MHBC_look_com_69690139345441
// MI455X (gfx1250) — compile-verified, matrix operand order checked by reading, not run on hardware
//
#include <hip/hip_runtime.h>
#include <hip/hip_bf16.h>

typedef __attribute__((ext_vector_type(16))) __bf16 v16bf;
typedef __attribute__((ext_vector_type(8)))  float  v8f;

// Problem constants
#define B_SZ   4
#define N_SZ   16384
#define DIM_   512
#define HEADS_ 8
#define DHEAD_ 64
#define INNER_ 512
#define WIN_   128
#define NW_    (N_SZ / WIN_)          // 128 windows
#define TOKENS (B_SZ * N_SZ)          // 65536
#define LN_EPS 1e-5f
#define SCALE_ 0.125f                  // DHEAD^-0.5

static __device__ __forceinline__ v8f wmma_bf16(v16bf a, v16bf b, v8f c) {
  // D = A(16x32 bf16) * B(32x16 bf16) + C(16x16 f32)
  return __builtin_amdgcn_wmma_f32_16x16x32_bf16(false, a, false, b, (short)0, c, false, false);
}

// ---------------------------------------------------------------------------
// Kernel 1: convert the four 512x512 weight matrices f32 -> bf16 (one pass)
// ---------------------------------------------------------------------------
__global__ __launch_bounds__(256) void wconv_kernel(
    const float* __restrict__ Wq, const float* __restrict__ Wk,
    const float* __restrict__ Wv, const float* __restrict__ Wo,
    __bf16* __restrict__ dq, __bf16* __restrict__ dk,
    __bf16* __restrict__ dv, __bf16* __restrict__ do_) {
  int m = blockIdx.y;
  const float* s = (m == 0) ? Wq : (m == 1) ? Wk : (m == 2) ? Wv : Wo;
  __bf16* d      = (m == 0) ? dq : (m == 1) ? dk : (m == 2) ? dv : do_;
  int i = blockIdx.x * 256 + threadIdx.x;   // grid.x = 1024 -> covers 262144
  d[i] = (__bf16)s[i];
}

// ---------------------------------------------------------------------------
// Kernel 2: fused QKV projection + LayerNorm (q,k) / plain (v)
//   grid.x = TOKENS/32 token tiles, grid.y = 3 (0:q from y, 1:k from x, 2:v)
//   block  = 256 threads = 8 waves; wave w owns output cols [64w, 64w+64)
//   Each wave: 32 tokens x 64 cols = 8 accumulators; B fragments reused 2x.
//   LayerNorm statistics computed directly from WMMA D-layout registers.
// ---------------------------------------------------------------------------
__global__ __launch_bounds__(256) void qkv_kernel(
    const float* __restrict__ x, const float* __restrict__ y,
    const __bf16* __restrict__ Wqb, const __bf16* __restrict__ Wkb,
    const __bf16* __restrict__ Wvb,
    const float* __restrict__ gq, const float* __restrict__ bq,
    const float* __restrict__ gk, const float* __restrict__ bk,
    __bf16* __restrict__ qo, __bf16* __restrict__ ko, __bf16* __restrict__ vo) {
  __shared__ alignas(32) __bf16 inA[32][528];   // 32 rows, padded (1056B = 33*32)
  __shared__ float redS[8][32];                 // per-wave row partial sums
  __shared__ float redQ[8][32];                 // per-wave row partial sumsq
  __shared__ float meanS[32], invS[32];

  const int msel = blockIdx.y;
  const float*  in = (msel == 0) ? y : x;
  const __bf16* W  = (msel == 0) ? Wqb : (msel == 1) ? Wkb : Wvb;
  const float*  g  = (msel == 0) ? gq : gk;
  const float*  bb = (msel == 0) ? bq : bk;
  __bf16* dst      = (msel == 0) ? qo : (msel == 1) ? ko : vo;

  const int tid = threadIdx.x;
  const int rowbase = blockIdx.x * 32;
  const size_t gbase = (size_t)rowbase * INNER_;

  // Stage 32x512 f32 rows into LDS as bf16 (float4 global loads)
  {
    const float4* src = (const float4*)(in + (size_t)rowbase * DIM_);
    for (int i = tid; i < 32 * 512 / 4; i += 256) {
      float4 f = src[i];
      int e = i * 4, r = e >> 9, c = e & 511;
      inA[r][c + 0] = (__bf16)f.x; inA[r][c + 1] = (__bf16)f.y;
      inA[r][c + 2] = (__bf16)f.z; inA[r][c + 3] = (__bf16)f.w;
    }
  }
  __syncthreads();

  const int wv = tid >> 5, lane = tid & 31;
  const int laneM = lane & 15, kh = lane >> 4;
  const int ncol0 = wv * 64;

  v8f acc[2][4] = {};
  for (int k0 = 0; k0 < 512; k0 += 32) {
    v16bf a0 = *(const v16bf*)&inA[laneM][k0 + kh * 16];
    v16bf a1 = *(const v16bf*)&inA[16 + laneM][k0 + kh * 16];
#pragma unroll
    for (int nt = 0; nt < 4; ++nt) {
      int n = ncol0 + nt * 16 + laneM;                 // output column = W row
      v16bf bf = *(const v16bf*)&W[(size_t)n * 512 + k0 + kh * 16];
      acc[0][nt] = wmma_bf16(a0, bf, acc[0][nt]);      // B reused for both M tiles
      acc[1][nt] = wmma_bf16(a1, bf, acc[1][nt]);
    }
  }

  // Preload per-lane gamma/beta (col = ncol0 + nt*16 + laneM)
  float gv[4], bv[4];
  if (msel < 2) {
#pragma unroll
    for (int nt = 0; nt < 4; ++nt) {
      int col = ncol0 + nt * 16 + laneM;
      gv[nt] = g[col]; bv[nt] = bb[col];
    }
    // Per-row partial sums straight from D layout:
    // acc VGPR r: lanes 0-15 = row mt*16+r, lanes 16-31 = row mt*16+r+8
#pragma unroll
    for (int mt = 0; mt < 2; ++mt)
#pragma unroll
      for (int r = 0; r < 8; ++r) {
        float s = 0.f, ss = 0.f;
#pragma unroll
        for (int nt = 0; nt < 4; ++nt) {
          float t = acc[mt][nt][r];
          s += t; ss += t * t;
        }
#pragma unroll
        for (int m = 8; m >= 1; m >>= 1) {   // reduce inside each 16-lane half
          s  += __shfl_xor(s,  m, 32);
          ss += __shfl_xor(ss, m, 32);
        }
        if (laneM == 0) {
          int row = mt * 16 + r + ((lane >= 16) ? 8 : 0);
          redS[wv][row] = s; redQ[wv][row] = ss;
        }
      }
    __syncthreads();
    if (tid < 32) {
      float s = 0.f, ss = 0.f;
#pragma unroll
      for (int w = 0; w < 8; ++w) { s += redS[w][tid]; ss += redQ[w][tid]; }
      float mean = s * (1.f / 512.f);
      float var  = ss * (1.f / 512.f) - mean * mean;
      meanS[tid] = mean;
      invS[tid]  = rsqrtf(var + LN_EPS);
    }
    __syncthreads();
    // Normalize in registers, store bf16
#pragma unroll
    for (int mt = 0; mt < 2; ++mt)
#pragma unroll
      for (int nt = 0; nt < 4; ++nt)
#pragma unroll
        for (int r = 0; r < 8; ++r) {
          int row = mt * 16 + ((lane < 16) ? r : r + 8);
          int col = ncol0 + nt * 16 + laneM;
          float t = (acc[mt][nt][r] - meanS[row]) * invS[row] * gv[nt] + bv[nt];
          dst[gbase + (size_t)row * INNER_ + col] = (__bf16)t;
        }
  } else {
    // v path: plain store
#pragma unroll
    for (int mt = 0; mt < 2; ++mt)
#pragma unroll
      for (int nt = 0; nt < 4; ++nt)
#pragma unroll
        for (int r = 0; r < 8; ++r) {
          int row = mt * 16 + ((lane < 16) ? r : r + 8);
          int col = ncol0 + nt * 16 + laneM;
          dst[gbase + (size_t)row * INNER_ + col] = (__bf16)acc[mt][nt][r];
        }
  }
}

// ---------------------------------------------------------------------------
// Kernel 3: windowed attention. One block per (window, head): 4096 blocks.
//   dots = q@k^T (16 WMMA/wave), in-register softmax on D layout,
//   attn -> d_out (f32) + LDS (bf16), out = attn@v (16 WMMA/wave).
// ---------------------------------------------------------------------------
__global__ __launch_bounds__(256) void attn_kernel(
    const __bf16* __restrict__ q, const __bf16* __restrict__ k,
    const __bf16* __restrict__ v,
    float* __restrict__ attn_out, __bf16* __restrict__ ao) {
  __shared__ alignas(32) unsigned char smem[59392];
  __bf16 (*qS)[80]  = (__bf16(*)[80])(smem);             // 128x64 pad80 : 20480B
  __bf16 (*kS)[80]  = (__bf16(*)[80])(smem + 20480);     // 128x64 pad80 : 20480B
  __bf16 (*aS)[144] = (__bf16(*)[144])(smem);            // 128x128 pad144: 36864B (overlaps q/k)
  __bf16 (*vT)[144] = (__bf16(*)[144])(smem + 40960);    // 64x128 pad144: 18432B

  const int tid = threadIdx.x;
  const int h  = blockIdx.x & (HEADS_ - 1);
  const int bw = blockIdx.x >> 3;
  const int tokbase = bw * WIN_;     // (b*NW + w)*128 == b*N + w*128

  // Stage q,k (uint4 = 8 bf16 per load), v transposed
  for (int c = tid; c < 1024; c += 256) {
    int row = c >> 3, col8 = (c & 7) * 8;
    size_t goff = (size_t)(tokbase + row) * INNER_ + h * DHEAD_ + col8;
    *(uint4*)&qS[row][col8] = *(const uint4*)&q[goff];
    *(uint4*)&kS[row][col8] = *(const uint4*)&k[goff];
  }
  for (int i = tid; i < WIN_ * DHEAD_; i += 256) {
    int row = i >> 6, col = i & 63;
    vT[col][row] = v[(size_t)(tokbase + row) * INNER_ + h * DHEAD_ + col];
  }
  __syncthreads();

  const int wv = tid >> 5, lane = tid & 31;
  const int laneM = lane & 15, kh = lane >> 4;
  const int m0 = wv * 16;

  // dots[m0..m0+16, 0..128] = q @ k^T
  v8f acc[8] = {};
#pragma unroll
  for (int k0 = 0; k0 < DHEAD_; k0 += 32) {
    v16bf a = *(const v16bf*)&qS[m0 + laneM][k0 + kh * 16];
#pragma unroll
    for (int nt = 0; nt < 8; ++nt) {
      v16bf bf = *(const v16bf*)&kS[nt * 16 + laneM][k0 + kh * 16];
      acc[nt] = wmma_bf16(a, bf, acc[nt]);
    }
  }

  // scale + softmax: xor 1/2/4/8 stays inside each 16-lane half == one row
#pragma unroll
  for (int r = 0; r < 8; ++r) {
    float mx = -3.4e38f;
#pragma unroll
    for (int nt = 0; nt < 8; ++nt) { acc[nt][r] *= SCALE_; mx = fmaxf(mx, acc[nt][r]); }
#pragma unroll
    for (int m = 8; m >= 1; m >>= 1) mx = fmaxf(mx, __shfl_xor(mx, m, 32));
    float sum = 0.f;
#pragma unroll
    for (int nt = 0; nt < 8; ++nt) { float e = __expf(acc[nt][r] - mx); acc[nt][r] = e; sum += e; }
#pragma unroll
    for (int m = 8; m >= 1; m >>= 1) sum += __shfl_xor(sum, m, 32);
    float rinv = 1.f / sum;
#pragma unroll
    for (int nt = 0; nt < 8; ++nt) acc[nt][r] *= rinv;
  }
  __syncthreads();  // q/k tiles dead; safe to overwrite with aS

  // attn -> global f32 (output #2) and bf16 -> LDS for attn@v
  const size_t abase = (size_t)(bw * HEADS_ + h) * (WIN_ * WIN_);
#pragma unroll
  for (int nt = 0; nt < 8; ++nt)
#pragma unroll
    for (int r = 0; r < 8; ++r) {
      int i = m0 + ((lane < 16) ? r : r + 8);
      int j = nt * 16 + laneM;
      float val = acc[nt][r];
      attn_out[abase + (size_t)i * WIN_ + j] = val;
      aS[i][j] = (__bf16)val;
    }
  __syncthreads();

  // out[m0..m0+16, 0..64] = attn @ v   (B[j][n] = v[j][n] = vT[n][j])
  v8f acc2[4] = {};
#pragma unroll
  for (int k0 = 0; k0 < WIN_; k0 += 32) {
    v16bf a = *(const v16bf*)&aS[m0 + laneM][k0 + kh * 16];
#pragma unroll
    for (int nt = 0; nt < 4; ++nt) {
      v16bf bf = *(const v16bf*)&vT[nt * 16 + laneM][k0 + kh * 16];
      acc2[nt] = wmma_bf16(a, bf, acc2[nt]);
    }
  }
#pragma unroll
  for (int nt = 0; nt < 4; ++nt)
#pragma unroll
    for (int r = 0; r < 8; ++r) {
      int i = m0 + ((lane < 16) ? r : r + 8);
      int n = nt * 16 + laneM;
      ao[(size_t)(tokbase + i) * INNER_ + h * DHEAD_ + n] = (__bf16)acc2[nt][r];
    }
}

// ---------------------------------------------------------------------------
// Kernel 4: y_new = y + attn_out @ Wout^T + bout   (f32 output #1)
//   32-token tiles, B fragments reused across 2 M tiles.
// ---------------------------------------------------------------------------
__global__ __launch_bounds__(256) void oproj_kernel(
    const __bf16* __restrict__ ao, const __bf16* __restrict__ Wob,
    const float* __restrict__ y, const float* __restrict__ bout,
    float* __restrict__ ynew) {
  __shared__ alignas(32) __bf16 inA[32][528];

  const int tid = threadIdx.x;
  const int rowbase = blockIdx.x * 32;

  // Stage 32 bf16 rows (already bf16 in ws)
  {
    const uint4* src = (const uint4*)(ao + (size_t)rowbase * INNER_);
    for (int c = tid; c < 2048; c += 256) {
      int row = c >> 6, ch = c & 63;
      *(uint4*)&inA[row][ch * 8] = src[(size_t)row * 64 + ch];
    }
  }
  __syncthreads();

  const int wv = tid >> 5, lane = tid & 31;
  const int laneM = lane & 15, kh = lane >> 4;
  const int ncol0 = wv * 64;

  v8f acc[2][4] = {};
  for (int k0 = 0; k0 < 512; k0 += 32) {
    v16bf a0 = *(const v16bf*)&inA[laneM][k0 + kh * 16];
    v16bf a1 = *(const v16bf*)&inA[16 + laneM][k0 + kh * 16];
#pragma unroll
    for (int nt = 0; nt < 4; ++nt) {
      int n = ncol0 + nt * 16 + laneM;
      v16bf bf = *(const v16bf*)&Wob[(size_t)n * 512 + k0 + kh * 16];
      acc[0][nt] = wmma_bf16(a0, bf, acc[0][nt]);
      acc[1][nt] = wmma_bf16(a1, bf, acc[1][nt]);
    }
  }
  float bv[4];
#pragma unroll
  for (int nt = 0; nt < 4; ++nt) bv[nt] = bout[ncol0 + nt * 16 + laneM];
#pragma unroll
  for (int mt = 0; mt < 2; ++mt)
#pragma unroll
    for (int nt = 0; nt < 4; ++nt)
#pragma unroll
      for (int r = 0; r < 8; ++r) {
        int row = mt * 16 + ((lane < 16) ? r : r + 8);
        int col = ncol0 + nt * 16 + laneM;
        size_t gi = (size_t)(rowbase + row) * DIM_ + col;
        ynew[gi] = acc[mt][nt][r] + y[gi] + bv[nt];
      }
}

// ---------------------------------------------------------------------------
extern "C" void kernel_launch(void* const* d_in, const int* in_sizes, int n_in,
                              void* d_out, int out_size, void* d_ws, size_t ws_size,
                              hipStream_t stream) {
  (void)in_sizes; (void)n_in; (void)out_size; (void)ws_size;
  const float* x    = (const float*)d_in[0];
  const float* y    = (const float*)d_in[1];
  const float* Wq   = (const float*)d_in[2];
  const float* gq   = (const float*)d_in[3];
  const float* bq   = (const float*)d_in[4];
  const float* Wk   = (const float*)d_in[5];
  const float* gk   = (const float*)d_in[6];
  const float* bk   = (const float*)d_in[7];
  const float* Wv   = (const float*)d_in[8];
  const float* Wout = (const float*)d_in[9];
  const float* bout = (const float*)d_in[10];

  float* ynew = (float*)d_out;
  float* attn = ynew + (size_t)B_SZ * N_SZ * DIM_;

  // Workspace layout (bytes)
  const size_t WSZ = (size_t)512 * 512 * sizeof(__bf16);           // 512KB per weight
  const size_t QSZ = (size_t)TOKENS * INNER_ * sizeof(__bf16);     // 64MB per tensor
  unsigned char* ws = (unsigned char*)d_ws;
  __bf16* Wqb = (__bf16*)(ws);
  __bf16* Wkb = (__bf16*)(ws + WSZ);
  __bf16* Wvb = (__bf16*)(ws + 2 * WSZ);
  __bf16* Wob = (__bf16*)(ws + 3 * WSZ);
  __bf16* qb  = (__bf16*)(ws + 4 * WSZ);
  __bf16* kb  = (__bf16*)(ws + 4 * WSZ + QSZ);
  __bf16* vb  = (__bf16*)(ws + 4 * WSZ + 2 * QSZ);
  __bf16* aob = (__bf16*)(ws + 4 * WSZ + 3 * QSZ);

  wconv_kernel<<<dim3(1024, 4), 256, 0, stream>>>(Wq, Wk, Wv, Wout, Wqb, Wkb, Wvb, Wob);
  qkv_kernel<<<dim3(TOKENS / 32, 3), 256, 0, stream>>>(x, y, Wqb, Wkb, Wvb,
                                                       gq, bq, gk, bk, qb, kb, vb);
  attn_kernel<<<dim3(B_SZ * NW_ * HEADS_), 256, 0, stream>>>(qb, kb, vb, attn, aob);
  oproj_kernel<<<dim3(TOKENS / 32), 256, 0, stream>>>(aob, Wob, y, bout, ynew);
}
